// MambaDecoder_44057774522970
// MI455X (gfx1250) — matrix operand order repaired and checked by reading, not compiled, not run on hardware
//
#include <hip/hip_runtime.h>
#include <hip/hip_bf16.h>
#include <cstdint>
#include <cstddef>

#define DIM    512
#define VOCAB  10000
#define DEPTH  3
#define DSTATE 16
#define DCONV  4
#define DINNER 1024
#define DTRANK 32
#define BB     4
#define LL     1024
#define CTXS   98
#define HEADS  8
#define ROWS   (BB*LL)              /* 4096 */
#define XDBLC  (DTRANK + 2*DSTATE)  /* 64 */
#define BKT    64                   /* GEMM k-tile */

typedef __attribute__((ext_vector_type(16))) __bf16 v16bf;
typedef __attribute__((ext_vector_type(8)))  __bf16 v8bf;
typedef __attribute__((ext_vector_type(8)))  float  v8f;

// ---------------- f32 -> bf16 (RNE) ----------------
__device__ __forceinline__ unsigned short f2bf(float f) {
    unsigned int u = __float_as_uint(f);
    u += 0x7fffu + ((u >> 16) & 1u);
    return (unsigned short)(u >> 16);
}

__global__ void f2bf_k(const float* __restrict__ src, unsigned short* __restrict__ dst, long long n) {
    long long i = (long long)blockIdx.x * blockDim.x + threadIdx.x;
    if (i < n) dst[i] = f2bf(src[i]);
}

// take first `cols` of each srcld-wide row, zero-pad to dstld
__global__ void f2bf_pad_k(const float* __restrict__ src, int srcld, int cols, int dstld,
                           unsigned short* __restrict__ dst, long long rows) {
    long long i = (long long)blockIdx.x * blockDim.x + threadIdx.x;
    if (i >= rows * dstld) return;
    long long r = i / dstld; int c = (int)(i % dstld);
    dst[i] = (c < cols) ? f2bf(src[r * srcld + c]) : (unsigned short)0;
}

// ---------------- WMMA GEMM: C[M,N] = A[M,K] * W[N,K]^T ----------------
// B tile: double-buffered in LDS via gfx1250 async global->LDS (ASYNCcnt).
// A fragments: software-pipelined in registers (loaded one k-tile ahead).
// All 8 B fragments of a tile preloaded -> one ds clause, back-to-back WMMAs.
__device__ __forceinline__ v16bf ldA(const unsigned short* row, int k0, int lhalf) {
    // contiguous A k order chosen here (not AMD's table order): lanes 0-15 hold K 0..15, lanes 16-31 hold K 16..31 (matches B fragments)
    v8bf lo = *(const v8bf*)(row + k0 + lhalf * 16);
    v8bf hi = *(const v8bf*)(row + k0 + lhalf * 16 + 8);
    v16bf r;
#pragma unroll
    for (int i = 0; i < 8; ++i) { r[i] = lo[i]; r[i + 8] = hi[i]; }
    return r;
}

// act: 0=none 1=softplus ; accum: C += result
__global__ __launch_bounds__(256)
void gemm_wmma_bf16_k(const unsigned short* __restrict__ A,
                      const unsigned short* __restrict__ W,
                      float* __restrict__ C,
                      int M, int N, int K,
                      const float* __restrict__ bias,
                      int act, int accum) {
    __shared__ __align__(32) unsigned short Ws[2][64 * BKT];   // 2 x 8KB

    const int tid   = threadIdx.x;
    const int lane  = tid & 31;
    const int wave  = tid >> 5;
    const int lrow  = lane & 15;
    const int lhalf = lane >> 4;
    const int mbase = blockIdx.y * 128 + wave * 16;
    const int nbase = blockIdx.x * 64;

    const unsigned long long wbase = (unsigned long long)(uintptr_t)W;
    const unsigned ldsbase = (unsigned)(uintptr_t)(&Ws[0][0]);   // flat->LDS offset (low 32 bits)

    // cooperative staging: 64 rows x 128B = 512 x 16B chunks, 2 chunks/thread
    auto stageTile = [&](int kt, int buf) {
#pragma unroll
        for (int j = 0; j < 2; ++j) {
            int ch  = tid * 2 + j;
            int row = ch >> 3;
            int kk  = (ch & 7) * 8;
            int gr  = min(nbase + row, N - 1);
            unsigned ga = (unsigned)(((size_t)gr * (size_t)K + (size_t)(kt * BKT + kk)) * 2u);
            unsigned la = ldsbase + (unsigned)((buf * 64 * BKT + row * BKT + kk) * 2);
            asm volatile("global_load_async_to_lds_b128 %0, %1, %2"
                         :: "v"(la), "v"(ga), "s"(wbase) : "memory");
        }
    };

    const unsigned short* Ar = A + (size_t)min(mbase + lrow, M - 1) * K;

    v8f acc[4] = {v8f{}, v8f{}, v8f{}, v8f{}};
    const int nk = K / BKT;

    stageTile(0, 0);
    v16bf a0 = ldA(Ar, 0, lhalf);             // A pipeline prologue
    v16bf a1 = ldA(Ar, 32, lhalf);

    for (int kt = 0; kt < nk; ++kt) {
        asm volatile("s_wait_asynccnt 0x0" ::: "memory");   // my async writes for this buf done
        __syncthreads();                                    // all waves staged; prior reads finished
        const int ktn = (kt + 1 < nk) ? (kt + 1) : (nk - 1);  // branchless tail (redundant restage)
        stageTile(ktn, (kt + 1) & 1);                        // overlaps with WMMAs below

        // preload all 8 B fragments for this tile (one ds clause, one wait)
        const int buf = kt & 1;
        const unsigned short* bb = &Ws[buf][0];
        v16bf b00 = *(const v16bf*)(bb + ( 0 + lrow) * BKT +  0 + lhalf * 16);
        v16bf b01 = *(const v16bf*)(bb + ( 0 + lrow) * BKT + 32 + lhalf * 16);
        v16bf b10 = *(const v16bf*)(bb + (16 + lrow) * BKT +  0 + lhalf * 16);
        v16bf b11 = *(const v16bf*)(bb + (16 + lrow) * BKT + 32 + lhalf * 16);
        v16bf b20 = *(const v16bf*)(bb + (32 + lrow) * BKT +  0 + lhalf * 16);
        v16bf b21 = *(const v16bf*)(bb + (32 + lrow) * BKT + 32 + lhalf * 16);
        v16bf b30 = *(const v16bf*)(bb + (48 + lrow) * BKT +  0 + lhalf * 16);
        v16bf b31 = *(const v16bf*)(bb + (48 + lrow) * BKT + 32 + lhalf * 16);

        // fetch NEXT tile's A fragments now; WMMAs below hide the latency
        __builtin_prefetch(Ar + ktn * BKT + BKT, 0, 1);
        v16bf na0 = ldA(Ar, ktn * BKT, lhalf);
        v16bf na1 = ldA(Ar, ktn * BKT + 32, lhalf);

        acc[0] = __builtin_amdgcn_wmma_f32_16x16x32_bf16(false, a0, false, b00, (short)0, acc[0], false, false);
        acc[1] = __builtin_amdgcn_wmma_f32_16x16x32_bf16(false, a0, false, b10, (short)0, acc[1], false, false);
        acc[2] = __builtin_amdgcn_wmma_f32_16x16x32_bf16(false, a0, false, b20, (short)0, acc[2], false, false);
        acc[3] = __builtin_amdgcn_wmma_f32_16x16x32_bf16(false, a0, false, b30, (short)0, acc[3], false, false);
        acc[0] = __builtin_amdgcn_wmma_f32_16x16x32_bf16(false, a1, false, b01, (short)0, acc[0], false, false);
        acc[1] = __builtin_amdgcn_wmma_f32_16x16x32_bf16(false, a1, false, b11, (short)0, acc[1], false, false);
        acc[2] = __builtin_amdgcn_wmma_f32_16x16x32_bf16(false, a1, false, b21, (short)0, acc[2], false, false);
        acc[3] = __builtin_amdgcn_wmma_f32_16x16x32_bf16(false, a1, false, b31, (short)0, acc[3], false, false);

        a0 = na0; a1 = na1;
    }

#pragma unroll
    for (int j = 0; j < 4; ++j) {
        int n = nbase + j * 16 + lrow;
        if (n >= N) continue;
        float bv = bias ? bias[n] : 0.0f;
#pragma unroll
        for (int r = 0; r < 8; ++r) {
            int m = mbase + lhalf * 8 + r;   // C/D layout: vgpr r -> M = r (+8 for upper lanes)
            if (m >= M) continue;
            float v = acc[j][r] + bv;
            if (act == 1) v = (v > 20.0f) ? v : __logf(1.0f + __expf(v));
            float* cp = C + (size_t)m * N + n;
            if (accum) v += *cp;
            *cp = v;
        }
    }
}

// ---------------- embedding ----------------
__global__ void embed_k(const int* __restrict__ ids, const float* __restrict__ tok,
                        const float* __restrict__ pos, float* __restrict__ h) {
    long long gid = (long long)blockIdx.x * blockDim.x + threadIdx.x;
    if (gid >= (long long)ROWS * DIM) return;
    int c = (int)(gid % DIM);
    long long t = gid / DIM;
    int l = (int)(t % LL);
    h[gid] = tok[(size_t)ids[t] * DIM + c] + pos[(size_t)l * DIM + c];
}

// ---------------- layernorm (one block per row) ----------------
__global__ void ln_k(const float* __restrict__ x, const float* __restrict__ g,
                     const float* __restrict__ b, float* __restrict__ y, int D) {
    __shared__ float s1[256], s2[256];
    int row = blockIdx.x, tid = threadIdx.x;
    const float* xr = x + (size_t)row * D;
    float a = 0.f, q = 0.f;
    for (int c = tid; c < D; c += blockDim.x) { float v = xr[c]; a += v; q += v * v; }
    s1[tid] = a; s2[tid] = q; __syncthreads();
    for (int o = 128; o > 0; o >>= 1) {
        if (tid < o) { s1[tid] += s1[tid + o]; s2[tid] += s2[tid + o]; }
        __syncthreads();
    }
    float mean = s1[0] / D;
    float var  = s2[0] / D - mean * mean;
    float inv  = rsqrtf(var + 1e-5f);
    for (int c = tid; c < D; c += blockDim.x)
        y[(size_t)row * D + c] = (xr[c] - mean) * inv * g[c] + b[c];
}

// ---------------- causal depthwise conv + SiLU (reads u = xz[:, :DINNER]) ----------------
__global__ void conv_silu_k(const float* __restrict__ xz, const float* __restrict__ cw,
                            const float* __restrict__ cb, float* __restrict__ uact) {
    long long gid = (long long)blockIdx.x * blockDim.x + threadIdx.x;
    if (gid >= (long long)ROWS * DINNER) return;
    int d = (int)(gid % DINNER);
    long long bl = gid / DINNER;
    int l = (int)(bl % LL);
    long long b = bl / LL;
    float acc = cb[d];
#pragma unroll
    for (int t = 0; t < DCONV; ++t) {
        int ls = l - (DCONV - 1) + t;
        if (ls >= 0) acc += cw[d * DCONV + t] * xz[(size_t)(b * LL + ls) * (2 * DINNER) + d];
    }
    uact[gid] = acc / (1.0f + __expf(-acc));
}

// ---------------- selective scan: one thread per (b, channel), h[16] in VGPRs ----------------
__global__ void scan_k(const float* __restrict__ delta, const float* __restrict__ uact,
                       const float* __restrict__ xdbl, const float* __restrict__ A_log,
                       const float* __restrict__ D_skip, float* __restrict__ y) {
    int t = blockIdx.x * blockDim.x + threadIdx.x;
    if (t >= BB * DINNER) return;
    int d = t % DINNER;
    int b = t / DINNER;
    float Arow[DSTATE], h[DSTATE];
#pragma unroll
    for (int s = 0; s < DSTATE; ++s) { Arow[s] = -__expf(A_log[d * DSTATE + s]); h[s] = 0.f; }
    float Dv = D_skip[d];
    for (int l = 0; l < LL; ++l) {
        size_t rb = (size_t)(b * LL + l);
        float dt = delta[rb * DINNER + d];
        float ut = uact[rb * DINNER + d];
        const float* xr = xdbl + rb * XDBLC;
        float du = dt * ut, acc = 0.f;
#pragma unroll
        for (int s = 0; s < DSTATE; ++s) {
            float dA = __expf(dt * Arow[s]);
            h[s] = dA * h[s] + du * xr[DTRANK + s];
            acc += h[s] * xr[DTRANK + DSTATE + s];
        }
        y[rb * DINNER + d] = acc + ut * Dv;
    }
}

// ---------------- y *= silu(z), z = xz[:, DINNER:2*DINNER] ----------------
__global__ void gate_k(const float* __restrict__ xz, float* __restrict__ y) {
    long long gid = (long long)blockIdx.x * blockDim.x + threadIdx.x;
    if (gid >= (long long)ROWS * DINNER) return;
    long long r = gid / DINNER; int c = (int)(gid % DINNER);
    float z = xz[r * (2 * DINNER) + DINNER + c];
    y[gid] *= z / (1.0f + __expf(-z));
}

// ---------------- attention scores + softmax: one wave per (b,h,l) row ----------------
__global__ void attn_softmax_k(const float* __restrict__ qkv, const float* __restrict__ kvctx,
                               float* __restrict__ attn) {
    int wid  = (int)((blockIdx.x * (long long)blockDim.x + threadIdx.x) >> 5);
    int lane = threadIdx.x & 31;
    if (wid >= BB * HEADS * LL) return;
    int l = wid % LL, hh = (wid / LL) % HEADS, b = wid / (LL * HEADS);
    const float* q = qkv + (size_t)(b * LL + l) * (3 * DIM) + hh * 64;
    float sc[4]; float mx = -1e30f;
#pragma unroll
    for (int j = 0; j < 4; ++j) {
        int s = lane + j * 32; float v = -1e30f;
        if (s < CTXS) {
            const float* kk = kvctx + (size_t)(b * CTXS + s) * (3 * DIM) + DIM + hh * 64;
            float acc = 0.f;
            for (int d = 0; d < 64; ++d) acc += q[d] * kk[d];
            v = acc * 0.125f;   // 1/sqrt(64)
        }
        sc[j] = v; mx = fmaxf(mx, v);
    }
    for (int o = 16; o > 0; o >>= 1) mx = fmaxf(mx, __shfl_xor(mx, o, 32));
    float sum = 0.f;
#pragma unroll
    for (int j = 0; j < 4; ++j) {
        int s = lane + j * 32;
        sc[j] = (s < CTXS) ? __expf(sc[j] - mx) : 0.f;
        sum += sc[j];
    }
    for (int o = 16; o > 0; o >>= 1) sum += __shfl_xor(sum, o, 32);
    float inv = 1.0f / sum;
#pragma unroll
    for (int j = 0; j < 4; ++j) {
        int s = lane + j * 32;
        if (s < CTXS) attn[(((size_t)b * HEADS + hh) * LL + l) * CTXS + s] = sc[j] * inv;
    }
}

// ---------------- attn mean over heads -> output slot ----------------
__global__ void attn_mean_k(const float* __restrict__ attn, float* __restrict__ out) {
    long long gid = (long long)blockIdx.x * blockDim.x + threadIdx.x;
    if (gid >= (long long)BB * LL * CTXS) return;
    int s = (int)(gid % CTXS);
    long long bl = gid / CTXS;
    int l = (int)(bl % LL); int b = (int)(bl / LL);
    float acc = 0.f;
#pragma unroll
    for (int h = 0; h < HEADS; ++h)
        acc += attn[(((size_t)b * HEADS + h) * LL + l) * CTXS + s];
    out[gid] = acc * (1.0f / HEADS);
}

// ---------------- attn @ V ----------------
__global__ void attn_apply_k(const float* __restrict__ attn, const float* __restrict__ kvctx,
                             float* __restrict__ ao) {
    long long gid = (long long)blockIdx.x * blockDim.x + threadIdx.x;
    if (gid >= (long long)ROWS * DIM) return;
    int d = (int)(gid % DIM);
    long long bl = gid / DIM;
    int l = (int)(bl % LL); int b = (int)(bl / LL);
    int hh = d >> 6;
    const float* ar = attn + (((size_t)b * HEADS + hh) * LL + l) * CTXS;
    float acc = 0.f;
    for (int s = 0; s < CTXS; ++s)
        acc += ar[s] * kvctx[(size_t)(b * CTXS + s) * (3 * DIM) + 2 * DIM + d];
    ao[gid] = acc;
}

// =======================================================================
extern "C" void kernel_launch(void* const* d_in, const int* in_sizes, int n_in,
                              void* d_out, int out_size, void* d_ws, size_t ws_size,
                              hipStream_t stream) {
    const int*   x_ids    = (const int*)  d_in[0];
    const float* context  = (const float*)d_in[1];
    const float* tok_emb  = (const float*)d_in[2];
    const float* pos_emb  = (const float*)d_in[3];
    const float* ln1_g    = (const float*)d_in[4];
    const float* ln1_b    = (const float*)d_in[5];
    const float* in_w     = (const float*)d_in[6];
    const float* conv_w   = (const float*)d_in[7];
    const float* conv_b   = (const float*)d_in[8];
    const float* xproj_w  = (const float*)d_in[9];
    const float* dt_w     = (const float*)d_in[10];
    const float* dt_b     = (const float*)d_in[11];
    const float* A_log    = (const float*)d_in[12];
    const float* D_skip   = (const float*)d_in[13];
    const float* mo_w     = (const float*)d_in[14];
    const float* ln2_g    = (const float*)d_in[15];
    const float* ln2_b    = (const float*)d_in[16];
    const float* ain_w    = (const float*)d_in[17];
    const float* ain_b    = (const float*)d_in[18];
    const float* aout_w   = (const float*)d_in[19];
    const float* aout_b   = (const float*)d_in[20];
    const float* lnf_g    = (const float*)d_in[21];
    const float* lnf_b    = (const float*)d_in[22];
    const float* logit_w  = (const float*)d_in[23];
    const float* logit_b  = (const float*)d_in[24];

    float* logits = (float*)d_out;                               // [4,1024,10000]
    float* attn_out_base = logits + (size_t)ROWS * VOCAB;        // 3 x [4,1024,98]

    // ---- workspace bump allocator ----
    char* wp = (char*)d_ws;
    auto alloc = [&](size_t bytes) -> void* {
        void* r = (void*)wp;
        wp += (bytes + 255) & ~(size_t)255;
        return r;
    };
    float*          h      = (float*)alloc((size_t)ROWS * DIM * 4);
    float*          hn     = (float*)alloc((size_t)ROWS * DIM * 4);
    unsigned short* hn16   = (unsigned short*)alloc((size_t)ROWS * DIM * 2);
    unsigned short* w16    = (unsigned short*)alloc((size_t)VOCAB * DIM * 2);   // max weight
    float*          xz     = (float*)alloc((size_t)ROWS * 2 * DINNER * 4);
    float*          uact   = (float*)alloc((size_t)ROWS * DINNER * 4);
    unsigned short* u16    = (unsigned short*)alloc((size_t)ROWS * DINNER * 2);
    float*          xdbl   = (float*)alloc((size_t)ROWS * XDBLC * 4);
    unsigned short* dt16   = (unsigned short*)alloc((size_t)ROWS * BKT * 2);    // K padded 32->64
    float*          delta  = (float*)alloc((size_t)ROWS * DINNER * 4);
    float*          yb     = (float*)alloc((size_t)ROWS * DINNER * 4);
    unsigned short* y16    = (unsigned short*)alloc((size_t)ROWS * DINNER * 2);
    unsigned short* ctx16  = (unsigned short*)alloc((size_t)BB * CTXS * DIM * 2);
    float*          kvctx  = (float*)alloc((size_t)BB * CTXS * 3 * DIM * 4);
    float*          qkv    = (float*)alloc((size_t)ROWS * 3 * DIM * 4);
    float*          attnb  = (float*)alloc((size_t)BB * HEADS * LL * CTXS * 4);
    float*          ao     = (float*)alloc((size_t)ROWS * DIM * 4);
    unsigned short* ao16   = (unsigned short*)alloc((size_t)ROWS * DIM * 2);

    auto cdiv = [](long long a, long long b) { return (int)((a + b - 1) / b); };
    auto conv = [&](const float* s, unsigned short* d, long long n) {
        f2bf_k<<<cdiv(n, 256), 256, 0, stream>>>(s, d, n);
    };
    auto gemm = [&](const unsigned short* A, const unsigned short* W, float* C,
                    int M, int N, int K, const float* bias, int act, int accum) {
        dim3 g(cdiv(N, 64), cdiv(M, 128));
        gemm_wmma_bf16_k<<<g, 256, 0, stream>>>(A, W, C, M, N, K, bias, act, accum);
    };

    // ---- embedding ----
    embed_k<<<cdiv((long long)ROWS * DIM, 256), 256, 0, stream>>>(x_ids, tok_emb, pos_emb, h);
    // ---- context -> bf16 (once) ----
    conv(context, ctx16, (long long)BB * CTXS * DIM);

    for (int i = 0; i < DEPTH; ++i) {
        // LN1
        ln_k<<<ROWS, 256, 0, stream>>>(h, ln1_g + i * DIM, ln1_b + i * DIM, hn, DIM);
        conv(hn, hn16, (long long)ROWS * DIM);
        // in_proj: xz = hn @ in_w^T
        conv(in_w + (size_t)i * 2 * DINNER * DIM, w16, (long long)2 * DINNER * DIM);
        gemm(hn16, w16, xz, ROWS, 2 * DINNER, DIM, nullptr, 0, 0);
        // depthwise conv + silu
        conv_silu_k<<<cdiv((long long)ROWS * DINNER, 256), 256, 0, stream>>>(
            xz, conv_w + (size_t)i * DINNER * DCONV, conv_b + (size_t)i * DINNER, uact);
        // x_proj
        conv(uact, u16, (long long)ROWS * DINNER);
        conv(xproj_w + (size_t)i * XDBLC * DINNER, w16, (long long)XDBLC * DINNER);
        gemm(u16, w16, xdbl, ROWS, XDBLC, DINNER, nullptr, 0, 0);
        // dt_proj + softplus (K padded 32 -> 64 so the k-tiled GEMM stays uniform)
        f2bf_pad_k<<<cdiv((long long)ROWS * BKT, 256), 256, 0, stream>>>(
            xdbl, XDBLC, DTRANK, BKT, dt16, ROWS);
        f2bf_pad_k<<<cdiv((long long)DINNER * BKT, 256), 256, 0, stream>>>(
            dt_w + (size_t)i * DINNER * DTRANK, DTRANK, DTRANK, BKT, w16, DINNER);
        gemm(dt16, w16, delta, ROWS, DINNER, BKT, dt_b + (size_t)i * DINNER, 1, 0);
        // selective scan: 4096 independent channel recurrences
        scan_k<<<(BB * DINNER) / 256, 256, 0, stream>>>(
            delta, uact, xdbl, A_log + (size_t)i * DINNER * DSTATE,
            D_skip + (size_t)i * DINNER, yb);
        // gate
        gate_k<<<cdiv((long long)ROWS * DINNER, 256), 256, 0, stream>>>(xz, yb);
        // out_proj, residual accumulate into h
        conv(yb, y16, (long long)ROWS * DINNER);
        conv(mo_w + (size_t)i * DIM * DINNER, w16, (long long)DIM * DINNER);
        gemm(y16, w16, h, ROWS, DIM, DINNER, nullptr, 0, 1);

        // LN2 + cross attention
        ln_k<<<ROWS, 256, 0, stream>>>(h, ln2_g + i * DIM, ln2_b + i * DIM, hn, DIM);
        conv(hn, hn16, (long long)ROWS * DIM);
        conv(ain_w + (size_t)i * 3 * DIM * DIM, w16, (long long)3 * DIM * DIM);
        gemm(hn16, w16, qkv, ROWS, 3 * DIM, DIM, ain_b + (size_t)i * 3 * DIM, 0, 0);
        gemm(ctx16, w16, kvctx, BB * CTXS, 3 * DIM, DIM, ain_b + (size_t)i * 3 * DIM, 0, 0);
        attn_softmax_k<<<(BB * HEADS * LL) / 8, 256, 0, stream>>>(qkv, kvctx, attnb);
        attn_mean_k<<<cdiv((long long)BB * LL * CTXS, 256), 256, 0, stream>>>(
            attnb, attn_out_base + (size_t)i * BB * LL * CTXS);
        attn_apply_k<<<cdiv((long long)ROWS * DIM, 256), 256, 0, stream>>>(attnb, kvctx, ao);
        conv(ao, ao16, (long long)ROWS * DIM);
        conv(aout_w + (size_t)i * DIM * DIM, w16, (long long)DIM * DIM);
        gemm(ao16, w16, h, ROWS, DIM, DIM, aout_b + (size_t)i * DIM, 0, 1);
    }

    // final LN + logits
    ln_k<<<ROWS, 256, 0, stream>>>(h, lnf_g, lnf_b, hn, DIM);
    conv(hn, hn16, (long long)ROWS * DIM);
    conv(logit_w, w16, (long long)VOCAB * DIM);
    gemm(hn16, w16, logits, ROWS, VOCAB, DIM, logit_b, 0, 0);
    (void)in_sizes; (void)n_in; (void)out_size; (void)ws_size;
}
